// Nettarj_Transformer_36301063586027
// MI455X (gfx1250) — compile-verified, matrix operand order checked by reading, not run on hardware
//
#include <hip/hip_runtime.h>

// ---------------- model constants ----------------
static constexpr int BATCH = 1024;
static constexpr int TSEQ  = 14;
static constexpr int NNODE = 50000;
static constexpr int MDIM  = 320;
static constexpr int DDIM  = 128;
static constexpr int DEG   = 16;
static constexpr int DM    = 768;
static constexpr int NH    = 8;
static constexpr int HD    = 96;       // DM/NH
static constexpr int FF    = 512;
static constexpr int NL    = 6;
static constexpr int SRC   = 10;
static constexpr int TGT   = 5;

typedef __attribute__((ext_vector_type(16))) __bf16 v16bf;
typedef __attribute__((ext_vector_type(8)))  float  v8f;

__device__ __forceinline__ unsigned short to_bf16u(float f) {
    unsigned int u = __float_as_uint(f);
    u += 0x7fffu + ((u >> 16) & 1u);          // round-to-nearest-even
    return (unsigned short)(u >> 16);
}

union bffrag { unsigned int u[8]; uint4 q[2]; v16bf v; };

// fragment = two contiguous 16B chunks: k-pairs {k0..k0+7} and {k0+16..k0+23} (per-lane +hi*8 folded into p)
__device__ __forceinline__ void load_frag(bffrag& f, const unsigned short* p, int k0) {
    f.q[0] = *(const uint4*)(p + k0);
    f.q[1] = *(const uint4*)(p + k0 + 16);
}

// ---------------- bf16 WMMA GEMM:  C[M,N] = A[M,K] * W[N,K]^T + bias ----------------
// one wave per 16x64 C strip (4 N-tiles): A fragment reused 4x, K pipelined 2-deep.
// All K used here are multiples of 64; all N multiples of 64.
__global__ void __launch_bounds__(256)
gemm_bf16_wmma(const unsigned short* __restrict__ A,
               const unsigned short* __restrict__ W,
               const float* __restrict__ bias,
               float* __restrict__ Cf, unsigned short* __restrict__ Cb,
               int Mrows, int Ncols, int Kdim, int relu)
{
    const int wave    = threadIdx.x >> 5;
    const int lane    = threadIdx.x & 31;
    const int stripsN = Ncols >> 6;
    const int strip   = blockIdx.x * 8 + wave;
    const int nStrips = (Mrows >> 4) * stripsN;
    if (strip >= nStrips) return;                // uniform per wave: EXEC stays all-1 for WMMA
    const int tm = strip / stripsN;
    const int tn = strip - tm * stripsN;
    const int row0 = tm << 4, col0 = tn << 6;

    const int ml = lane & 15;                    // A row / B column within tile
    const int hi = lane >> 4;                    // K half-select per ISA 16-bit layout

    const unsigned short* arow  = A + (size_t)(row0 + ml) * Kdim + (hi << 3);
    const unsigned short* wbase = W + (size_t)(col0 + ml) * Kdim + (hi << 3);
    const size_t wstep = (size_t)16 * Kdim;      // next 16-column subtile

    v8f acc0 = {}, acc1 = {}, acc2 = {}, acc3 = {};
    bffrag aA, bA0, bA1, bA2, bA3;               // set A (even k-steps)
    bffrag aB, bB0, bB1, bB2, bB3;               // set B (odd k-steps)

    // prologue: load set A at k=0
    load_frag(aA, arow, 0);
    load_frag(bA0, wbase,             0);
    load_frag(bA1, wbase + wstep,     0);
    load_frag(bA2, wbase + 2 * wstep, 0);
    load_frag(bA3, wbase + 3 * wstep, 0);

    for (int k0 = 0; k0 < Kdim; k0 += 64) {
        // launch set B loads (k0+32), then consume set A
        load_frag(aB, arow, k0 + 32);
        load_frag(bB0, wbase,             k0 + 32);
        load_frag(bB1, wbase + wstep,     k0 + 32);
        load_frag(bB2, wbase + 2 * wstep, k0 + 32);
        load_frag(bB3, wbase + 3 * wstep, k0 + 32);
        __builtin_prefetch(arow + k0 + 64, 0, 1);
        acc0 = __builtin_amdgcn_wmma_f32_16x16x32_bf16(false, aA.v, false, bA0.v, (short)0, acc0, false, false);
        acc1 = __builtin_amdgcn_wmma_f32_16x16x32_bf16(false, aA.v, false, bA1.v, (short)0, acc1, false, false);
        acc2 = __builtin_amdgcn_wmma_f32_16x16x32_bf16(false, aA.v, false, bA2.v, (short)0, acc2, false, false);
        acc3 = __builtin_amdgcn_wmma_f32_16x16x32_bf16(false, aA.v, false, bA3.v, (short)0, acc3, false, false);
        // launch next set A loads (k0+64) unless last step, then consume set B
        if (k0 + 64 < Kdim) {
            load_frag(aA, arow, k0 + 64);
            load_frag(bA0, wbase,             k0 + 64);
            load_frag(bA1, wbase + wstep,     k0 + 64);
            load_frag(bA2, wbase + 2 * wstep, k0 + 64);
            load_frag(bA3, wbase + 3 * wstep, k0 + 64);
        }
        acc0 = __builtin_amdgcn_wmma_f32_16x16x32_bf16(false, aB.v, false, bB0.v, (short)0, acc0, false, false);
        acc1 = __builtin_amdgcn_wmma_f32_16x16x32_bf16(false, aB.v, false, bB1.v, (short)0, acc1, false, false);
        acc2 = __builtin_amdgcn_wmma_f32_16x16x32_bf16(false, aB.v, false, bB2.v, (short)0, acc2, false, false);
        acc3 = __builtin_amdgcn_wmma_f32_16x16x32_bf16(false, aB.v, false, bB3.v, (short)0, acc3, false, false);
    }

    // C 16x16 f32 layout: lane -> N = lane&15 ; VGPR r -> M = (lane>=16 ? 8 : 0) + r
    const int mrow = row0 + (hi << 3);
    #pragma unroll
    for (int j = 0; j < 4; ++j) {
        const v8f acc = (j == 0) ? acc0 : (j == 1) ? acc1 : (j == 2) ? acc2 : acc3;
        const int n = col0 + (j << 4) + ml;
        const float bv = bias ? bias[n] : 0.0f;
        #pragma unroll
        for (int r = 0; r < 8; ++r) {
            float v = acc[r] + bv;
            if (relu) v = fmaxf(v, 0.0f);
            const size_t idx = (size_t)(mrow + r) * Ncols + n;
            if (Cf) Cf[idx] = v;
            if (Cb) Cb[idx] = to_bf16u(v);
        }
    }
}

// ---------------- small conversion / build kernels ----------------
__global__ void cvt_bf16_kernel(const float* __restrict__ s, unsigned short* __restrict__ d, size_t n)
{
    size_t i = (size_t)blockIdx.x * blockDim.x + threadIdx.x;
    if (i < n) d[i] = to_bf16u(s[i]);
}

// gat_w is [K=M, N=M]; store transposed [N, K] in bf16 so every GEMM is A * W^T
__global__ void gatw_transpose_kernel(const float* __restrict__ src, unsigned short* __restrict__ dst)
{
    int idx = blockIdx.x * blockDim.x + threadIdx.x;
    if (idx >= MDIM * MDIM) return;
    int n = idx / MDIM, k = idx - n * MDIM;
    dst[idx] = to_bf16u(src[k * MDIM + n]);
}

__global__ void build_h_kernel(const float* __restrict__ w, const float* __restrict__ b,
                               unsigned short* __restrict__ H)
{
    size_t idx = (size_t)blockIdx.x * blockDim.x + threadIdx.x;
    if (idx >= (size_t)NNODE * MDIM) return;
    int k = (int)(idx % MDIM);
    H[idx] = to_bf16u(w[idx] + b[k]);
}

// per-node attention scores:  ai = WH . a[0:M],  aj = WH . a[M:2M]
__global__ void gat_scores_kernel(const float* __restrict__ WH, const float* __restrict__ ga,
                                  float* __restrict__ ai, float* __restrict__ aj)
{
    int n = blockIdx.x * blockDim.x + threadIdx.x;
    if (n >= NNODE) return;
    const float* w = WH + (size_t)n * MDIM;
    float a = 0.f, b = 0.f;
    for (int k = 0; k < MDIM; ++k) { float t = w[k]; a += t * ga[k]; b += t * ga[MDIM + k]; }
    ai[n] = a; aj[n] = b;
}

// one block per (b,t): leaky-relu scores over DEG, softmax, elu(weighted sum of WH rows)
__global__ void __launch_bounds__(64)
gat_aggregate_kernel(const int* __restrict__ node_ids, const int* __restrict__ neighbors,
                     const float* __restrict__ WH, const float* __restrict__ ai,
                     const float* __restrict__ aj, float* __restrict__ wh)
{
    const int bt  = blockIdx.x;                 // 0 .. B*T-1
    const int nid = node_ids[bt];
    __shared__ float alpha[DEG];
    __shared__ int   nbr[DEG];
    const int tid = threadIdx.x;
    if (tid < DEG) {
        int nb = neighbors[(size_t)nid * DEG + tid];
        nbr[tid] = nb;
        float e = ai[nid] + aj[nb];
        alpha[tid] = (e > 0.f) ? e : 0.2f * e;  // leaky_relu(0.2)
    }
    __syncthreads();
    if (tid == 0) {
        float mx = -1e30f;
        for (int d = 0; d < DEG; ++d) mx = fmaxf(mx, alpha[d]);
        float s = 0.f;
        for (int d = 0; d < DEG; ++d) { alpha[d] = __expf(alpha[d] - mx); s += alpha[d]; }
        float inv = 1.0f / s;
        for (int d = 0; d < DEG; ++d) alpha[d] *= inv;
    }
    __syncthreads();
    for (int m = tid; m < MDIM; m += 64) {
        float acc = 0.f;
        #pragma unroll
        for (int d = 0; d < DEG; ++d) acc += alpha[d] * WH[(size_t)nbr[d] * MDIM + m];
        wh[(size_t)bt * MDIM + m] = (acc > 0.f) ? acc : (__expf(acc) - 1.0f);  // elu
    }
}

// build encoder/decoder input: concat(node_emb, dir_emb, wh) + positional encoding
__global__ void build_x_kernel(const int* __restrict__ node_ids, const int* __restrict__ dir_ids,
                               const float* __restrict__ node_w, const float* __restrict__ node_b,
                               const float* __restrict__ dir_w, const float* __restrict__ dir_b,
                               const float* __restrict__ wh,
                               float* __restrict__ Xf, unsigned short* __restrict__ Xb,
                               int L, int tstart)
{
    size_t idx = (size_t)blockIdx.x * blockDim.x + threadIdx.x;
    size_t total = (size_t)L * BATCH * DM;
    if (idx >= total) return;
    const int c   = (int)(idx % DM);
    const size_t row = idx / DM;          // pos*B + b
    const int b   = (int)(row % BATCH);
    const int pos = (int)(row / BATCH);
    const int t   = tstart + pos;
    float v;
    if (c < MDIM)
        v = node_w[(size_t)node_ids[b * TSEQ + t] * MDIM + c] + node_b[c];
    else if (c < MDIM + DDIM)
        v = dir_w[(size_t)dir_ids[b * TSEQ + t] * DDIM + (c - MDIM)] + dir_b[c - MDIM];
    else
        v = wh[((size_t)b * TSEQ + t) * MDIM + (c - MDIM - DDIM)];
    const int i2 = c & ~1;
    const float div = __expf(-9.210340371976184f * (float)i2 / (float)DM); // ln(10000)
    const float ang = (float)pos * div;
    v += (c & 1) ? __cosf(ang) : __sinf(ang);
    Xf[idx] = v;
    Xb[idx] = to_bf16u(v);
}

// LayerNorm over 768 cols, optional residual; wave32 per row with shuffle reductions
__global__ void __launch_bounds__(256)
ln_kernel(const float* __restrict__ X, const float* __restrict__ R,
          const float* __restrict__ g, const float* __restrict__ bb,
          float* __restrict__ Of, unsigned short* __restrict__ Obf, int rows)
{
    const int wave = threadIdx.x >> 5, lane = threadIdx.x & 31;
    const int row  = blockIdx.x * 8 + wave;
    if (row >= rows) return;
    const float* x = X + (size_t)row * DM;
    const float* r = R ? R + (size_t)row * DM : nullptr;
    float v[24]; float s = 0.f;
    #pragma unroll
    for (int i = 0; i < 24; ++i) {
        const int c = lane + i * 32;
        float t = x[c] + (r ? r[c] : 0.0f);
        v[i] = t; s += t;
    }
    #pragma unroll
    for (int o = 16; o; o >>= 1) s += __shfl_xor(s, o, 32);
    const float mu = s * (1.0f / DM);
    float var = 0.f;
    #pragma unroll
    for (int i = 0; i < 24; ++i) { float d = v[i] - mu; var += d * d; }
    #pragma unroll
    for (int o = 16; o; o >>= 1) var += __shfl_xor(var, o, 32);
    const float rs = rsqrtf(var * (1.0f / DM) + 1e-5f);
    #pragma unroll
    for (int i = 0; i < 24; ++i) {
        const int c = lane + i * 32;
        const float y = (v[i] - mu) * rs * g[c] + bb[c];
        if (Of)  Of[(size_t)row * DM + c]  = y;
        if (Obf) Obf[(size_t)row * DM + c] = to_bf16u(y);
    }
}

// scalar attention: one thread per (qpos, b, h). rows are pos*B + b; heads contiguous.
__global__ void attention_kernel(const float* __restrict__ Q, int qstride, int qoff,
                                 const float* __restrict__ KV, int kvstride, int koff, int voff,
                                 unsigned short* __restrict__ Ob,
                                 int Lq, int Lk, int causal)
{
    const int idx = blockIdx.x * blockDim.x + threadIdx.x;
    if (idx >= BATCH * NH * Lq) return;
    const int h  = idx & (NH - 1);
    const int b  = (idx >> 3) & (BATCH - 1);
    const int qp = idx >> 13;                       // / (NH*BATCH)
    const float* q = Q + (size_t)(qp * BATCH + b) * qstride + qoff + h * HD;

    float s[SRC];                                   // Lk <= 10
    float mx = -1e30f;
    for (int j = 0; j < Lk; ++j) {
        const float* k = KV + (size_t)(j * BATCH + b) * kvstride + koff + h * HD;
        float d = 0.f;
        for (int c = 0; c < HD; ++c) d += q[c] * k[c];
        d *= 0.10206207261596577f;                  // 1/sqrt(96)
        if (causal && j > qp) d = -1e9f;
        s[j] = d; mx = fmaxf(mx, d);
    }
    float ssum = 0.f;
    for (int j = 0; j < Lk; ++j) { s[j] = __expf(s[j] - mx); ssum += s[j]; }
    const float inv = 1.0f / ssum;
    unsigned short* o = Ob + (size_t)(qp * BATCH + b) * DM + h * HD;
    for (int c = 0; c < HD; ++c) {
        float acc = 0.f;
        for (int j = 0; j < Lk; ++j)
            acc += s[j] * inv * KV[(size_t)(j * BATCH + b) * kvstride + voff + h * HD + c];
        o[c] = to_bf16u(acc);
    }
}

// output: [5, B, 800] = concat(y_norm, taxi_emb)
__global__ void write_out_kernel(const float* __restrict__ y, const int* __restrict__ taxi_ids,
                                 const float* __restrict__ tw, const float* __restrict__ tb,
                                 float* __restrict__ out)
{
    size_t idx = (size_t)blockIdx.x * blockDim.x + threadIdx.x;
    size_t total = (size_t)TGT * BATCH * 800;
    if (idx >= total) return;
    const int c = (int)(idx % 800);
    const size_t row = idx / 800;                   // t*B + b
    const int b = (int)(row % BATCH);
    if (c < DM) out[idx] = y[row * DM + c];
    else        out[idx] = tw[(size_t)taxi_ids[b] * 32 + (c - DM)] + tb[c - DM];
}

// ---------------- host orchestration ----------------
extern "C" void kernel_launch(void* const* d_in, const int* in_sizes, int n_in,
                              void* d_out, int out_size, void* d_ws, size_t ws_size,
                              hipStream_t stream)
{
    const int*   node_ids  = (const int*)  d_in[0];
    const int*   dir_ids   = (const int*)  d_in[1];
    const int*   taxi_ids  = (const int*)  d_in[2];
    const int*   neighbors = (const int*)  d_in[3];
    const float* node_w    = (const float*)d_in[4];
    const float* node_b    = (const float*)d_in[5];
    const float* dir_w     = (const float*)d_in[6];
    const float* dir_b     = (const float*)d_in[7];
    const float* taxi_w    = (const float*)d_in[8];
    const float* taxi_b    = (const float*)d_in[9];
    const float* gat_w     = (const float*)d_in[10];
    const float* gat_a     = (const float*)d_in[11];
    const float* enc_qkv_w = (const float*)d_in[12];
    const float* enc_qkv_b = (const float*)d_in[13];
    const float* enc_out_w = (const float*)d_in[14];
    const float* enc_out_b = (const float*)d_in[15];
    const float* enc_ln1_g = (const float*)d_in[16];
    const float* enc_ln1_b = (const float*)d_in[17];
    const float* enc_ln2_g = (const float*)d_in[18];
    const float* enc_ln2_b = (const float*)d_in[19];
    const float* enc_ff1_w = (const float*)d_in[20];
    const float* enc_ff1_b = (const float*)d_in[21];
    const float* enc_ff2_w = (const float*)d_in[22];
    const float* enc_ff2_b = (const float*)d_in[23];
    const float* enc_nrm_g = (const float*)d_in[24];
    const float* enc_nrm_b = (const float*)d_in[25];
    const float* dec_qkv_w = (const float*)d_in[26];
    const float* dec_qkv_b = (const float*)d_in[27];
    const float* dec_out_w = (const float*)d_in[28];
    const float* dec_out_b = (const float*)d_in[29];
    const float* dec_xqkv_w= (const float*)d_in[30];
    const float* dec_xqkv_b= (const float*)d_in[31];
    const float* dec_xout_w= (const float*)d_in[32];
    const float* dec_xout_b= (const float*)d_in[33];
    const float* dec_ln1_g = (const float*)d_in[34];
    const float* dec_ln1_b = (const float*)d_in[35];
    const float* dec_ln2_g = (const float*)d_in[36];
    const float* dec_ln2_b = (const float*)d_in[37];
    const float* dec_ln3_g = (const float*)d_in[38];
    const float* dec_ln3_b = (const float*)d_in[39];
    const float* dec_ff1_w = (const float*)d_in[40];
    const float* dec_ff1_b = (const float*)d_in[41];
    const float* dec_ff2_w = (const float*)d_in[42];
    const float* dec_ff2_b = (const float*)d_in[43];
    const float* dec_nrm_g = (const float*)d_in[44];
    const float* dec_nrm_b = (const float*)d_in[45];
    (void)in_sizes; (void)n_in; (void)out_size; (void)ws_size;

    // ---- carve workspace ----
    unsigned char* wp = (unsigned char*)d_ws;
    auto carve = [&](size_t bytes) -> void* {
        void* r = (void*)wp; wp += (bytes + 255) & ~(size_t)255; return r;
    };
    const size_t QKV_W = (size_t)NL * 3 * DM * DM;
    const size_t OUT_W = (size_t)NL * DM * DM;
    const size_t FF1_W = (size_t)NL * FF * DM;
    const size_t FF2_W = (size_t)NL * DM * FF;

    unsigned short* w_gatT = (unsigned short*)carve((size_t)MDIM * MDIM * 2);
    unsigned short* w_eqkv = (unsigned short*)carve(QKV_W * 2);
    unsigned short* w_eout = (unsigned short*)carve(OUT_W * 2);
    unsigned short* w_eff1 = (unsigned short*)carve(FF1_W * 2);
    unsigned short* w_eff2 = (unsigned short*)carve(FF2_W * 2);
    unsigned short* w_dqkv = (unsigned short*)carve(QKV_W * 2);
    unsigned short* w_dout = (unsigned short*)carve(OUT_W * 2);
    unsigned short* w_dxqkv= (unsigned short*)carve(QKV_W * 2);
    unsigned short* w_dxout= (unsigned short*)carve(OUT_W * 2);
    unsigned short* w_dff1 = (unsigned short*)carve(FF1_W * 2);
    unsigned short* w_dff2 = (unsigned short*)carve(FF2_W * 2);

    unsigned short* Hb  = (unsigned short*)carve((size_t)NNODE * MDIM * 2);
    float* WH  = (float*)carve((size_t)NNODE * MDIM * 4);
    float* ai_ = (float*)carve((size_t)NNODE * 4);
    float* aj_ = (float*)carve((size_t)NNODE * 4);
    float* whb = (float*)carve((size_t)BATCH * TSEQ * MDIM * 4);

    const size_t ER = (size_t)SRC * BATCH;            // 10240 encoder rows
    const size_t DR = (size_t)TGT * BATCH;            // 5120 decoder rows
    float*          Xf  = (float*)carve(ER * DM * 4);
    unsigned short* Xb  = (unsigned short*)carve(ER * DM * 2);
    float*          Yf  = (float*)carve(DR * DM * 4);
    unsigned short* Yb  = (unsigned short*)carve(DR * DM * 2);
    unsigned short* Mb  = (unsigned short*)carve(ER * DM * 2);
    float*          qkvb= (float*)carve(ER * 3 * DM * 4);
    unsigned short* Ob  = (unsigned short*)carve(ER * DM * 2);
    float*          Af  = (float*)carve(ER * DM * 4);
    unsigned short* F1b = (unsigned short*)carve(ER * FF * 2);
    float*          KVf = (float*)carve(ER * 2 * DM * 4);
    float*          YQf = (float*)carve(DR * DM * 4);

    auto conv = [&](const float* s, unsigned short* d, size_t n) {
        cvt_bf16_kernel<<<(unsigned)((n + 255) / 256), 256, 0, stream>>>(s, d, n);
    };
    auto gemm = [&](const unsigned short* A, const unsigned short* W, const float* bias,
                    float* Cf, unsigned short* Cb, int Mr, int Nc, int Kd, int relu) {
        const int strips = (Mr / 16) * (Nc / 64);     // one wave per 16x64 strip
        gemm_bf16_wmma<<<(strips + 7) / 8, 256, 0, stream>>>(A, W, bias, Cf, Cb, Mr, Nc, Kd, relu);
    };
    auto ln = [&](const float* X, const float* R, const float* g, const float* b,
                  float* Of, unsigned short* Obf, int rows) {
        ln_kernel<<<(rows + 7) / 8, 256, 0, stream>>>(X, R, g, b, Of, Obf, rows);
    };

    // ---- weight conversion (fp32 -> bf16) ----
    gatw_transpose_kernel<<<(MDIM * MDIM + 255) / 256, 256, 0, stream>>>(gat_w, w_gatT);
    conv(enc_qkv_w, w_eqkv, QKV_W);  conv(enc_out_w, w_eout, OUT_W);
    conv(enc_ff1_w, w_eff1, FF1_W);  conv(enc_ff2_w, w_eff2, FF2_W);
    conv(dec_qkv_w, w_dqkv, QKV_W);  conv(dec_out_w, w_dout, OUT_W);
    conv(dec_xqkv_w, w_dxqkv, QKV_W);conv(dec_xout_w, w_dxout, OUT_W);
    conv(dec_ff1_w, w_dff1, FF1_W);  conv(dec_ff2_w, w_dff2, FF2_W);

    // ---- GAT: WHall = (table + b) @ gat_w over all unique nodes, then gather ----
    build_h_kernel<<<(unsigned)(((size_t)NNODE * MDIM + 255) / 256), 256, 0, stream>>>(node_w, node_b, Hb);
    gemm(Hb, w_gatT, nullptr, WH, nullptr, NNODE, MDIM, MDIM, 0);
    gat_scores_kernel<<<(NNODE + 255) / 256, 256, 0, stream>>>(WH, gat_a, ai_, aj_);
    gat_aggregate_kernel<<<BATCH * TSEQ, 64, 0, stream>>>(node_ids, neighbors, WH, ai_, aj_, whb);

    // ---- build encoder / decoder inputs (+PE) ----
    build_x_kernel<<<(unsigned)((ER * DM + 255) / 256), 256, 0, stream>>>(
        node_ids, dir_ids, node_w, node_b, dir_w, dir_b, whb, Xf, Xb, SRC, 0);
    build_x_kernel<<<(unsigned)((DR * DM + 255) / 256), 256, 0, stream>>>(
        node_ids, dir_ids, node_w, node_b, dir_w, dir_b, whb, Yf, Yb, TGT, SRC - 1);

    // ---- encoder ----
    for (int l = 0; l < NL; ++l) {
        gemm(Xb, w_eqkv + (size_t)l * 3 * DM * DM, enc_qkv_b + l * 3 * DM,
             qkvb, nullptr, (int)ER, 3 * DM, DM, 0);
        attention_kernel<<<(BATCH * NH * SRC + 255) / 256, 256, 0, stream>>>(
            qkvb, 3 * DM, 0, qkvb, 3 * DM, DM, 2 * DM, Ob, SRC, SRC, 0);
        gemm(Ob, w_eout + (size_t)l * DM * DM, enc_out_b + l * DM, Af, nullptr, (int)ER, DM, DM, 0);
        ln(Af, Xf, enc_ln1_g + l * DM, enc_ln1_b + l * DM, Xf, Xb, (int)ER);
        gemm(Xb, w_eff1 + (size_t)l * FF * DM, enc_ff1_b + l * FF, nullptr, F1b, (int)ER, FF, DM, 1);
        gemm(F1b, w_eff2 + (size_t)l * DM * FF, enc_ff2_b + l * DM, Af, nullptr, (int)ER, DM, FF, 0);
        ln(Af, Xf, enc_ln2_g + l * DM, enc_ln2_b + l * DM, Xf, Xb, (int)ER);
    }
    ln(Xf, nullptr, enc_nrm_g, enc_nrm_b, nullptr, Mb, (int)ER);   // mem (bf16 for cross-KV GEMMs)

    // ---- decoder ----
    for (int l = 0; l < NL; ++l) {
        gemm(Yb, w_dqkv + (size_t)l * 3 * DM * DM, dec_qkv_b + l * 3 * DM,
             qkvb, nullptr, (int)DR, 3 * DM, DM, 0);
        attention_kernel<<<(BATCH * NH * TGT + 255) / 256, 256, 0, stream>>>(
            qkvb, 3 * DM, 0, qkvb, 3 * DM, DM, 2 * DM, Ob, TGT, TGT, 1);
        gemm(Ob, w_dout + (size_t)l * DM * DM, dec_out_b + l * DM, Af, nullptr, (int)DR, DM, DM, 0);
        ln(Af, Yf, dec_ln1_g + l * DM, dec_ln1_b + l * DM, Yf, Yb, (int)DR);
        // cross-attention: Q from y, KV from mem
        gemm(Yb, w_dxqkv + (size_t)l * 3 * DM * DM, dec_xqkv_b + l * 3 * DM,
             YQf, nullptr, (int)DR, DM, DM, 0);
        gemm(Mb, w_dxqkv + (size_t)l * 3 * DM * DM + (size_t)DM * DM, dec_xqkv_b + l * 3 * DM + DM,
             KVf, nullptr, (int)ER, 2 * DM, DM, 0);
        attention_kernel<<<(BATCH * NH * TGT + 255) / 256, 256, 0, stream>>>(
            YQf, DM, 0, KVf, 2 * DM, 0, DM, Ob, TGT, SRC, 0);
        gemm(Ob, w_dxout + (size_t)l * DM * DM, dec_xout_b + l * DM, Af, nullptr, (int)DR, DM, DM, 0);
        ln(Af, Yf, dec_ln2_g + l * DM, dec_ln2_b + l * DM, Yf, Yb, (int)DR);
        gemm(Yb, w_dff1 + (size_t)l * FF * DM, dec_ff1_b + l * FF, nullptr, F1b, (int)DR, FF, DM, 1);
        gemm(F1b, w_dff2 + (size_t)l * DM * FF, dec_ff2_b + l * DM, Af, nullptr, (int)DR, DM, FF, 0);
        ln(Af, Yf, dec_ln3_g + l * DM, dec_ln3_b + l * DM, Yf, Yb, (int)DR);
    }
    ln(Yf, nullptr, dec_nrm_g, dec_nrm_b, Af, nullptr, (int)DR);   // final decoder norm -> Af

    // ---- concat with taxi embedding ----
    write_out_kernel<<<(unsigned)(((size_t)TGT * BATCH * 800 + 255) / 256), 256, 0, stream>>>(
        Af, taxi_ids, taxi_w, taxi_b, (float*)d_out);
}
